// GraphAttn_19335942767154
// MI455X (gfx1250) — matrix operand order repaired and checked by reading, not compiled, not run on hardware
//
#include <hip/hip_runtime.h>
#include <hip/hip_bf16.h>

typedef __bf16 bf16_t;
typedef __attribute__((ext_vector_type(16))) __bf16 v16bf;
typedef __attribute__((ext_vector_type(8)))  __bf16 v8bf;
typedef __attribute__((ext_vector_type(8)))  float  v8f;

#define B_SZ   512
#define N_SZ   128
#define F_SZ   256
#define H_SZ   8
#define D_SZ   32
#define OUT_SZ 256

// ---------- WMMA fragment loaders (CDNA5 ISA 7.12.2 layouts, wave32) ----------
// All loads are two contiguous 16-byte chunks per lane.

__device__ __forceinline__ v16bf combine(v8bf lo, v8bf hi) {
  return __builtin_shufflevector(lo, hi, 0,1,2,3,4,5,6,7,8,9,10,11,12,13,14,15);
}

// A fragment: 16x32 (MxK) bf16 from row-major src, leading dim ld (elements).
// lane l: row = l&15 ; K = {kh..kh+7, 16+kh..16+kh+7}, kh = 8*(l>>4)
__device__ __forceinline__ v16bf ldA(const bf16_t* base, int ld, int lane) {
  const bf16_t* r = base + (lane & 15) * ld + ((lane >> 4) << 3);
  const v8bf lo = *(const v8bf*)(r);
  const v8bf hi = *(const v8bf*)(r + 16);
  return combine(lo, hi);
}

// B fragment (32x16 KxN) where B = srcT, src row-major [N][K], leading dim ld.
// lane l: col = l&15 ; K = {kh..kh+7, 16+kh..16+kh+7}, kh = 8*(l>>4)
__device__ __forceinline__ v16bf ldBt(const bf16_t* bt, int ld, int lane) {
  const bf16_t* r = bt + (lane & 15) * ld + ((lane >> 4) << 3);
  const v8bf lo = *(const v8bf*)(r);
  const v8bf hi = *(const v8bf*)(r + 16);
  return combine(lo, hi);
}

__device__ __forceinline__ v8bf pack8(const v8f a) {
  v8bf p;
#pragma unroll
  for (int j = 0; j < 8; ++j) p[j] = (bf16_t)a[j];
  return p;
}

__device__ __forceinline__ float fast_tanh(float v) {
#if __has_builtin(__builtin_amdgcn_tanhf)
  return __builtin_amdgcn_tanhf(v);
#elif __has_builtin(__builtin_amdgcn_tanh_f32)
  return __builtin_amdgcn_tanh_f32(v);
#else
  return tanhf(v);
#endif
}

#define WMMA_BF16(A, Bf, C) \
  __builtin_amdgcn_wmma_f32_16x16x32_bf16(false, (A), false, (Bf), (short)0, (C), false, false)

// ---------------- Kernel 0: one-time weight transpose/convert ----------------
// wqkvt: [3][H][D][F] bf16 ; wot: [OUT][OUT] bf16 with wot[n][k] = wo[k][n]
__global__ __launch_bounds__(256) void gat_prep(
    const float* __restrict__ wq, const float* __restrict__ wk,
    const float* __restrict__ wv, const float* __restrict__ wo,
    bf16_t* __restrict__ wqkvt, bf16_t* __restrict__ wot) {
  const int id = blockIdx.x * 256 + threadIdx.x;
  const int NW = 3 * H_SZ * F_SZ * D_SZ;  // 196608
  if (id < NW) {
    const int q = id / (H_SZ * F_SZ * D_SZ);
    const int r = id % (H_SZ * F_SZ * D_SZ);
    const int h = r >> 13;          // / (F*D)
    const int f = (r >> 5) & 255;
    const int d = r & 31;
    const float* w = (q == 0) ? wq : (q == 1) ? wk : wv;
    wqkvt[((q * H_SZ + h) * D_SZ + d) * F_SZ + f] = (bf16_t)w[r];
  } else {
    const int r = id - NW;
    if (r < OUT_SZ * OUT_SZ) {
      const int k = r >> 8;
      const int n = r & 255;
      wot[n * OUT_SZ + k] = (bf16_t)wo[r];
    }
  }
}

// ---------------- Kernel 1: per-(b,h) fused attention head ----------------
__global__ __launch_bounds__(256) void gat_head(
    const float* __restrict__ x,  const float* __restrict__ adj,
    const bf16_t* __restrict__ wqkvt, bf16_t* __restrict__ hcat) {
  __shared__ __align__(16) bf16_t lX [N_SZ * F_SZ];      // 64KB; reused as S[128][128]
  __shared__ __align__(16) bf16_t lWt[3 * D_SZ * F_SZ];  // 48KB  (Wq/Wk/Wv^T, [q][d][f])
  __shared__ __align__(16) bf16_t lQ [N_SZ * D_SZ];      // 8KB   Q [n][d]
  __shared__ __align__(16) bf16_t lK [N_SZ * D_SZ];      // 8KB   K [n][d]
  __shared__ __align__(16) bf16_t lVt[D_SZ * N_SZ];      // 8KB   V^T [d][n]

  const int b    = blockIdx.x >> 3;
  const int h    = blockIdx.x & 7;
  const int tid  = threadIdx.x;
  const int w    = tid >> 5;          // wave 0..7: owns node-tile w
  const int lane = tid & 31;

  // ---- Phase 1: stage X[b] (fp32->bf16, packed stores) and W^T (plain copy) ----
  const float* xb = x + (size_t)b * (N_SZ * F_SZ);
  for (int i = tid; i < (N_SZ * F_SZ) / 8; i += 256) {
    const float4 f0 = ((const float4*)xb)[2 * i];
    const float4 f1 = ((const float4*)xb)[2 * i + 1];
    v8bf p;
    p[0] = (bf16_t)f0.x; p[1] = (bf16_t)f0.y; p[2] = (bf16_t)f0.z; p[3] = (bf16_t)f0.w;
    p[4] = (bf16_t)f1.x; p[5] = (bf16_t)f1.y; p[6] = (bf16_t)f1.z; p[7] = (bf16_t)f1.w;
    ((v8bf*)lX)[i] = p;
  }
  for (int i = tid; i < (3 * D_SZ * F_SZ) / 8; i += 256) {  // 3072 uint4
    const int q = i >> 10;
    const int j = i & 1023;
    ((uint4*)lWt)[i] = ((const uint4*)wqkvt)[(q * H_SZ + h) * 1024 + j];
  }
  __syncthreads();

  // ---- Phase 2: QKV projections, register-blocked: wave w owns node-tile w ----
  // Q,K transposed orientation: P^T = W^T * X^T (4 tiles share B = X^T)
  // V   normal orientation:     P   = X * W     (2 tiles share A = X)
  {
    v8f accQ0 = {}, accQ1 = {}, accK0 = {}, accK1 = {}, accV0 = {}, accV1 = {};
    const bf16_t* xrow = lX + w * 16 * F_SZ;
#pragma unroll
    for (int k = 0; k < 8; ++k) {
      const v16bf bX  = ldBt(xrow + k * 32, F_SZ, lane);
      const v16bf aX  = ldA (xrow + k * 32, F_SZ, lane);
      const v16bf aQ0 = ldA (lWt + 0 * 4096 + k * 32, F_SZ, lane);  // Wq^T, d 0..15
      const v16bf aQ1 = ldA (lWt + 1 * 4096 + k * 32, F_SZ, lane);  // Wq^T, d 16..31
      const v16bf aK0 = ldA (lWt + 2 * 4096 + k * 32, F_SZ, lane);  // Wk^T, d 0..15
      const v16bf aK1 = ldA (lWt + 3 * 4096 + k * 32, F_SZ, lane);  // Wk^T, d 16..31
      const v16bf bV0 = ldBt(lWt + 4 * 4096 + k * 32, F_SZ, lane);  // Wv^T, d 0..15
      const v16bf bV1 = ldBt(lWt + 5 * 4096 + k * 32, F_SZ, lane);  // Wv^T, d 16..31
      accQ0 = WMMA_BF16(aQ0, bX, accQ0);
      accQ1 = WMMA_BF16(aQ1, bX, accQ1);
      accK0 = WMMA_BF16(aK0, bX, accK0);
      accK1 = WMMA_BF16(aK1, bX, accK1);
      accV0 = WMMA_BF16(aX, bV0, accV0);
      accV1 = WMMA_BF16(aX, bV1, accV1);
    }
    const int r8 = (lane >> 4) << 3;
    const int n  = w * 16 + (lane & 15);          // node (lane col for Q/K tiles)
    *(v8bf*)(lQ + n * D_SZ + r8)      = pack8(accQ0);
    *(v8bf*)(lQ + n * D_SZ + 16 + r8) = pack8(accQ1);
    *(v8bf*)(lK + n * D_SZ + r8)      = pack8(accK0);
    *(v8bf*)(lK + n * D_SZ + 16 + r8) = pack8(accK1);
    const int dv = lane & 15;                     // d (lane col for V tiles)
    const int n0 = w * 16 + r8;                   // node rows (contig over j)
    *(v8bf*)(lVt + dv * N_SZ + n0)        = pack8(accV0);
    *(v8bf*)(lVt + (16 + dv) * N_SZ + n0) = pack8(accV1);
  }
  __syncthreads();

  // ---- Phase 3: S^T-tiles = K * Q^T, masked tanh, store S row-major [q][key] ----
  // Wave w owns key-tile w: A fragment hoisted out of the query loop.
  bf16_t* lS = lX;  // alias; X no longer needed
  const float* adjb = adj + (size_t)b * (N_SZ * N_SZ);
  const float scale = 0.17677669529663687f;  // 1/sqrt(32)
  {
    const v16bf a   = ldA(lK + w * 16 * D_SZ, D_SZ, lane);
    const int   kr0 = w * 16 + ((lane >> 4) << 3);   // key rows (contig over j)
    for (int nq = 0; nq < 8; ++nq) {
      const v16bf bb = ldBt(lQ + nq * 16 * D_SZ, D_SZ, lane);
      v8f acc = {};
      acc = WMMA_BF16(a, bb, acc);
      const int qc = nq * 16 + (lane & 15);          // query index (lane col)
      const float4 a0 = *(const float4*)(adjb + qc * N_SZ + kr0);
      const float4 a1 = *(const float4*)(adjb + qc * N_SZ + kr0 + 4);
      v8f sv;
      sv[0] = acc[0] * a0.x; sv[1] = acc[1] * a0.y; sv[2] = acc[2] * a0.z; sv[3] = acc[3] * a0.w;
      sv[4] = acc[4] * a1.x; sv[5] = acc[5] * a1.y; sv[6] = acc[6] * a1.z; sv[7] = acc[7] * a1.w;
#pragma unroll
      for (int j = 0; j < 8; ++j) sv[j] = fast_tanh(sv[j] * scale);
      *(v8bf*)(lS + qc * N_SZ + kr0) = pack8(sv);
    }
  }
  __syncthreads();

  // ---- Phase 4: Hh^T-tiles = V^T * S^T ; wave w owns query-tile w ----
  {
    v8f acc0 = {}, acc1 = {};
    const bf16_t* srow = lS + w * 16 * N_SZ;
#pragma unroll
    for (int k = 0; k < 4; ++k) {
      const v16bf bS = ldBt(srow + k * 32, N_SZ, lane);            // shared B
      const v16bf a0 = ldA(lVt + k * 32, N_SZ, lane);              // d 0..15
      const v16bf a1 = ldA(lVt + 16 * N_SZ + k * 32, N_SZ, lane);  // d 16..31
      acc0 = WMMA_BF16(a0, bS, acc0);
      acc1 = WMMA_BF16(a1, bS, acc1);
    }
    const int qc = w * 16 + (lane & 15);
    const int r8 = (lane >> 4) << 3;
    bf16_t* dst = hcat + ((size_t)b * N_SZ + qc) * OUT_SZ + h * D_SZ;
    *(v8bf*)(dst + r8)      = pack8(acc0);
    *(v8bf*)(dst + 16 + r8) = pack8(acc1);
  }
}

// ---------------- Kernel 2: out = relu(Hcat @ Wo), computed as Wo^T * H^T ----------------
__global__ __launch_bounds__(256) void gat_out(
    const bf16_t* __restrict__ hcat, const bf16_t* __restrict__ wot,
    float* __restrict__ out) {
  __shared__ __align__(16) bf16_t lWot[OUT_SZ * OUT_SZ];  // 128KB, [n][k] = wo[k][n]
  const int tid  = threadIdx.x;
  const int w    = tid >> 5;          // wave 0..7: owns row-tile w
  const int lane = tid & 31;
  const size_t rows0 = (size_t)blockIdx.x * 128;

  for (int i = tid; i < (OUT_SZ * OUT_SZ) / 8; i += 256)   // 8192 uint4
    ((uint4*)lWot)[i] = ((const uint4*)wot)[i];
  __syncthreads();

  // Cache all 8 H^T B-fragments for this wave's 16 rows (read global ONCE).
  v16bf bH[8];
#pragma unroll
  for (int k = 0; k < 8; ++k)
    bH[k] = ldBt(hcat + (rows0 + w * 16) * OUT_SZ + k * 32, OUT_SZ, lane);

  const size_t r  = rows0 + w * 16 + (lane & 15);   // global row (lane col)
  const int    r8 = (lane >> 4) << 3;
  for (int mog = 0; mog < 4; ++mog) {               // 4 groups of 4 output-col tiles
    v8f acc0 = {}, acc1 = {}, acc2 = {}, acc3 = {};
    const bf16_t* wbase = lWot + (mog * 4) * 16 * OUT_SZ;
#pragma unroll
    for (int k = 0; k < 8; ++k) {
      const v16bf a0 = ldA(wbase + 0 * 16 * OUT_SZ + k * 32, OUT_SZ, lane);
      const v16bf a1 = ldA(wbase + 1 * 16 * OUT_SZ + k * 32, OUT_SZ, lane);
      const v16bf a2 = ldA(wbase + 2 * 16 * OUT_SZ + k * 32, OUT_SZ, lane);
      const v16bf a3 = ldA(wbase + 3 * 16 * OUT_SZ + k * 32, OUT_SZ, lane);
      acc0 = WMMA_BF16(a0, bH[k], acc0);
      acc1 = WMMA_BF16(a1, bH[k], acc1);
      acc2 = WMMA_BF16(a2, bH[k], acc2);
      acc3 = WMMA_BF16(a3, bH[k], acc3);
    }
    v8f accs[4] = {acc0, acc1, acc2, acc3};
#pragma unroll
    for (int i = 0; i < 4; ++i) {
      const int o0 = (mog * 4 + i) * 16 + r8;       // output cols (contig over j)
      float4 f0, f1;
      f0.x = fmaxf(accs[i][0], 0.f); f0.y = fmaxf(accs[i][1], 0.f);
      f0.z = fmaxf(accs[i][2], 0.f); f0.w = fmaxf(accs[i][3], 0.f);
      f1.x = fmaxf(accs[i][4], 0.f); f1.y = fmaxf(accs[i][5], 0.f);
      f1.z = fmaxf(accs[i][6], 0.f); f1.w = fmaxf(accs[i][7], 0.f);
      *(float4*)(out + r * OUT_SZ + o0)     = f0;
      *(float4*)(out + r * OUT_SZ + o0 + 4) = f1;
    }
  }
}

extern "C" void kernel_launch(void* const* d_in, const int* in_sizes, int n_in,
                              void* d_out, int out_size, void* d_ws, size_t ws_size,
                              hipStream_t stream) {
  const float* x   = (const float*)d_in[0];
  const float* adj = (const float*)d_in[1];
  const float* wq  = (const float*)d_in[2];
  const float* wk  = (const float*)d_in[3];
  const float* wv  = (const float*)d_in[4];
  const float* wo  = (const float*)d_in[5];

  // Workspace layout (bytes):
  //   [0, 33554432)              hcat  [B][N][OUT] bf16
  //   [33554432, 33947648)       wqkvt [3][H][D][F] bf16
  //   [33947648, 34078720)       wot   [OUT][OUT] bf16 (transposed)
  char* ws = (char*)d_ws;
  bf16_t* hcat  = (bf16_t*)(ws);
  bf16_t* wqkvt = (bf16_t*)(ws + (size_t)B_SZ * N_SZ * OUT_SZ * 2);
  bf16_t* wot   = (bf16_t*)(ws + (size_t)B_SZ * N_SZ * OUT_SZ * 2 + 3 * H_SZ * F_SZ * D_SZ * 2);

  gat_prep<<<dim3(1024), dim3(256), 0, stream>>>(wq, wk, wv, wo, wqkvt, wot);
  gat_head<<<dim3(B_SZ * H_SZ), dim3(256), 0, stream>>>(x, adj, wqkvt, hcat);
  gat_out <<<dim3((B_SZ * N_SZ) / 128), dim3(256), 0, stream>>>(hcat, wot, (float*)d_out);
}
